// GNNClassifier_57011395887296
// MI455X (gfx1250) — matrix operand order repaired and checked by reading, not compiled, not run on hardware
//
#include <hip/hip_runtime.h>
#include <hip/hip_bf16.h>

// ---- problem constants (match reference) ----
#define Nn      100000
#define Ee      1600000
#define Hh      128
#define Rr      4
#define Bb      4
#define Ll      3
#define Gg      128
#define FEATd   6
#define NCLS    2
#define EPSf    1e-5f
#define KTOT    640          // (R+1)*H
#define KCHUNKS 20           // KTOT / 32
#define A_STRIDE 648         // 640 + 8 bf16 pad -> breaks LDS bank conflicts

typedef __attribute__((ext_vector_type(16))) __bf16 v16bf;
typedef __attribute__((ext_vector_type(8)))  float  v8f;

__device__ __forceinline__ unsigned short f2bf(float f) {
  unsigned int u = __float_as_uint(f);
  u += 0x7FFFu + ((u >> 16) & 1u);          // round-to-nearest-even
  return (unsigned short)(u >> 16);
}

__device__ __forceinline__ void atomicMaxF(float* addr, float val) {
  int* ai = (int*)addr;
  int old = *ai;
  while (__int_as_float(old) < val) {
    int assumed = old;
    old = atomicCAS(ai, assumed, __float_as_int(val));
    if (old == assumed) break;
  }
}

// ---------------- utility ----------------
__global__ void fill_f32(float* p, float v, long n) {
  long i = (long)blockIdx.x * blockDim.x + threadIdx.x;
  long stride = (long)gridDim.x * blockDim.x;
  for (; i < n; i += stride) p[i] = v;
}

// ---------------- encoder: x = emb[x_ids] + relu(node_feat@feat_w+b) + virtual ----------------
__global__ __launch_bounds__(256) void encode_kernel(
    const int* __restrict__ x_ids, const float* __restrict__ nfeat,
    const float* __restrict__ emb, const float* __restrict__ featw,
    const float* __restrict__ featb, const float* __restrict__ virt,
    float* __restrict__ x) {
  int i = blockIdx.x * 256 + threadIdx.x;      // over N*H
  int c = i & (Hh - 1);
  int n = i >> 7;
  if (n >= Nn) return;
  float acc = featb[c];
#pragma unroll
  for (int f = 0; f < FEATd; ++f) acc += nfeat[n * FEATd + f] * featw[f * Hh + c];
  acc = fmaxf(acc, 0.f);
  x[i] = emb[(long)x_ids[n] * Hh + c] + acc + virt[c];
}

// ---------------- per-(dst,rel) edge counts -> 1/max(cnt,1) (layer-invariant) ----------------
__global__ void count_edges(const int* __restrict__ dst, const int* __restrict__ et,
                            float* __restrict__ inv) {
  int e = blockIdx.x * blockDim.x + threadIdx.x;
  if (e >= Ee) return;
  atomicAdd(inv + (long)dst[e] * Rr + et[e], 1.0f);
}
__global__ void invert_counts(float* inv) {
  int i = blockIdx.x * blockDim.x + threadIdx.x;
  if (i < Nn * Rr) inv[i] = 1.0f / fmaxf(inv[i], 1.0f);
}

// ---------------- edge scatter: S[dst*R+et] += x[src]  (one wave32 per edge, float4 lanes) ----------------
__global__ __launch_bounds__(256) void scatter_edges(
    const float* __restrict__ x, const int* __restrict__ src,
    const int* __restrict__ dst, const int* __restrict__ et,
    float* __restrict__ S) {
  int e = blockIdx.x * 8 + (threadIdx.x >> 5);
  if (e >= Ee) return;
  int lane = threadIdx.x & 31;
  long sa  = (long)src[e] * Hh + lane * 4;
  long key = (long)dst[e] * Rr + et[e];
  const float4 v = *(const float4*)(x + sa);
  float* p = S + key * Hh + lane * 4;
  atomicAdd(p + 0, v.x); atomicAdd(p + 1, v.y);
  atomicAdd(p + 2, v.z); atomicAdd(p + 3, v.w);
}

// ---------------- pre-pack Wcat=[W_0..W_3;root] into exact per-lane B-fragment order (bf16) ----------------
// B order chosen here (contiguous k, not AMD's table order; 32x16): lane -> n = lane%16, half = lane/16; element j -> K = j + 16*half.
__global__ void build_bpack(const float* __restrict__ bases, const float* __restrict__ comp,
                            const float* __restrict__ root, int layer,
                            unsigned short* __restrict__ Bpack) {
  int idx = blockIdx.x * blockDim.x + threadIdx.x;     // 8*20*32*16 = 81920
  if (idx >= 8 * KCHUNKS * 32 * 16) return;
  int j    = idx & 15;
  int lane = (idx >> 4) & 31;
  int kb   = (idx >> 9) % KCHUNKS;
  int wv   = idx / (KCHUNKS * 32 * 16);
  int n    = (wv << 4) + (lane & 15);
  int half = lane >> 4;
  int K    = (kb << 5) + j + (half << 4);
  float val;
  if (K < Rr * Hh) {
    int r = K >> 7, k = K & 127;
    const float* cp = comp + ((long)layer * Rr + r) * Bb;
    float s = 0.f;
#pragma unroll
    for (int b = 0; b < Bb; ++b)
      s += cp[b] * bases[(((long)layer * Bb + b) * Hh + k) * Hh + n];
    val = s;
  } else {
    val = root[((long)layer * Hh + (K - Rr * Hh)) * Hh + n];
  }
  Bpack[idx] = f2bf(val);
}

// ---------------- WMMA GEMM: h[N,128] = [S_norm | x] (N x 640) @ Wcat (640 x 128) + rbias ----------------
// Block = 16 rows x 128 cols; 8 waves, one 16x16 tile each; K accumulated in 20 chunks of 32 (bf16).
__global__ __launch_bounds__(256) void rgcn_gemm(
    const float* __restrict__ S, const float* __restrict__ inv,
    const float* __restrict__ x, const unsigned short* __restrict__ Bpack,
    const float* __restrict__ rbias_l, float* __restrict__ hout) {
  __shared__ unsigned short Abf[16 * A_STRIDE];
  const int m0  = blockIdx.x << 4;
  const int tid = threadIdx.x;

  // Stage A-strip into LDS as bf16, folding the per-(dst,rel) mean normalization.
  for (int i = tid; i < 16 * KTOT; i += 256) {
    int row = i / KTOT, c = i - row * KTOT;
    int n = m0 + row;
    float v = 0.f;
    if (n < Nn) {
      if (c < Rr * Hh) {
        long key = (long)n * Rr + (c >> 7);
        v = S[key * Hh + (c & 127)] * inv[key];
      } else {
        v = x[(long)n * Hh + (c - Rr * Hh)];
      }
    }
    Abf[row * A_STRIDE + c] = f2bf(v);
  }
  __syncthreads();

  const int wv = tid >> 5, lane = tid & 31;
  const int nloc = lane & 15, half = lane >> 4;

  v8f acc;
  {
    float rb = rbias_l[(wv << 4) + nloc];     // bias depends only on output channel
#pragma unroll
    for (int i = 0; i < 8; ++i) acc[i] = rb;
  }

  const unsigned short* Arow = Abf + nloc * A_STRIDE;   // A: m = lane%16
#pragma unroll
  for (int kb = 0; kb < KCHUNKS; ++kb) {
    union U16 { v16bf v; unsigned short s[16]; } a, b;
    const int kbase = kb << 5;
    // A fragment: element j -> K = kbase + 16*half + j (matches B pack)
#pragma unroll
    for (int j = 0; j < 8; ++j) a.s[j]     = Arow[kbase + (half << 4) + j];
#pragma unroll
    for (int j = 0; j < 8; ++j) a.s[8 + j] = Arow[kbase + (half << 4) + 8 + j];
    // B fragment: pre-packed, 32 contiguous bytes per lane
    const unsigned short* bp = Bpack + ((((wv * KCHUNKS) + kb) * 32 + lane) << 4);
#pragma unroll
    for (int j = 0; j < 16; ++j) b.s[j] = bp[j];
    acc = __builtin_amdgcn_wmma_f32_16x16x32_bf16(false, a.v, false, b.v,
                                                  (short)0, acc, false, false);
  }
  // C/D layout: VGPR i -> M = i + 8*half, N = lane%16
#pragma unroll
  for (int i = 0; i < 8; ++i) {
    int m = m0 + i + (half << 3);
    if (m < Nn) hout[(long)m * Hh + (wv << 4) + nloc] = acc[i];
  }
}

// ---------------- BatchNorm: accumulate sum/sumsq, then residual ReLU update ----------------
__global__ __launch_bounds__(256) void bn_accum(const float* __restrict__ h,
                                                float* __restrict__ stats) {
  __shared__ float sh[2 * 256];
  int c = threadIdx.x & 127;
  int sub = threadIdx.x >> 7;
  float s = 0.f, q = 0.f;
  int base = blockIdx.x * 128;
  for (int r = base + sub; r < base + 128 && r < Nn; r += 2) {
    float v = h[(long)r * Hh + c];
    s += v; q += v * v;
  }
  sh[threadIdx.x]       = s;
  sh[256 + threadIdx.x] = q;
  __syncthreads();
  if (sub == 0) {
    atomicAdd(stats + c,      s + sh[threadIdx.x + 128]);
    atomicAdd(stats + Hh + c, q + sh[256 + threadIdx.x + 128]);
  }
}

__global__ __launch_bounds__(256) void bn_update(
    const float* __restrict__ h, const float* __restrict__ stats,
    const float* __restrict__ gamma_l, const float* __restrict__ beta_l,
    float* __restrict__ x) {
  int i = blockIdx.x * 256 + threadIdx.x;
  int c = i & 127;
  if ((i >> 7) >= Nn) return;
  const float invN = 1.0f / (float)Nn;
  float mu  = stats[c] * invN;
  float var = stats[Hh + c] * invN - mu * mu;
  float hn  = gamma_l[c] * (h[i] - mu) * rsqrtf(var + EPSf) + beta_l[c];
  x[i] += fmaxf(hn, 0.f);
}

// ---------------- readout: mean + max pooling per graph ----------------
__global__ __launch_bounds__(256) void pool_kernel(
    const float* __restrict__ x, const int* __restrict__ batch,
    float* __restrict__ psum, float* __restrict__ pmax) {
  int i = blockIdx.x * 256 + threadIdx.x;
  int n = i >> 7, c = i & 127;
  if (n >= Nn) return;
  int g = batch[n];
  float v = x[i];
  atomicAdd(psum + g * Hh + c, v);
  atomicMaxF(pmax + g * Hh + c, v);
}
__global__ void graph_counts(const int* __restrict__ batch, float* __restrict__ gcnt) {
  int n = blockIdx.x * blockDim.x + threadIdx.x;
  if (n < Nn) atomicAdd(gcnt + batch[n], 1.0f);
}

// ---------------- head MLP ----------------
__global__ __launch_bounds__(256) void mlp1_kernel(
    const float* __restrict__ psum, const float* __restrict__ pmax,
    const float* __restrict__ gcnt, const float* __restrict__ w1,
    const float* __restrict__ b1, float* __restrict__ ghid) {
  int i = blockIdx.x * 256 + threadIdx.x;   // G*H
  int g = i >> 7, j = i & 127;
  if (g >= Gg) return;
  float cnt  = gcnt[g];
  float icnt = 1.0f / fmaxf(cnt, 1.0f);
  bool  nz   = cnt > 0.f;
  float acc  = b1[j];
  for (int k = 0; k < Hh; ++k) acc += psum[g * Hh + k] * icnt * w1[k * Hh + j];
  for (int k = 0; k < Hh; ++k) {
    float mx = nz ? pmax[g * Hh + k] : 0.f;
    acc += mx * w1[(Hh + k) * Hh + j];
  }
  ghid[i] = fmaxf(acc, 0.f);
}
__global__ void mlp2_kernel(const float* __restrict__ ghid, const float* __restrict__ w2,
                            const float* __restrict__ b2, float* __restrict__ out) {
  int i = blockIdx.x * blockDim.x + threadIdx.x;   // G*NCLS = 256
  if (i >= Gg * NCLS) return;
  int g = i / NCLS, c = i % NCLS;
  float acc = b2[c];
  for (int j = 0; j < Hh; ++j) acc += ghid[g * Hh + j] * w2[j * NCLS + c];
  out[i] = acc;   // temperature = max(1.0, 1e-4) = 1.0
}

// ---------------- host orchestration ----------------
extern "C" void kernel_launch(void* const* d_in, const int* in_sizes, int n_in,
                              void* d_out, int out_size, void* d_ws, size_t ws_size,
                              hipStream_t stream) {
  (void)in_sizes; (void)n_in; (void)out_size; (void)ws_size;
  const int*   x_ids = (const int*)d_in[0];
  const int*   eidx  = (const int*)d_in[1];
  const int*   etype = (const int*)d_in[2];
  const int*   batch = (const int*)d_in[3];
  const float* nfeat = (const float*)d_in[4];
  const float* emb   = (const float*)d_in[5];
  const float* featw = (const float*)d_in[6];
  const float* featb = (const float*)d_in[7];
  const float* virt  = (const float*)d_in[8];
  const float* bases = (const float*)d_in[9];
  const float* comp  = (const float*)d_in[10];
  const float* root  = (const float*)d_in[11];
  const float* rbias = (const float*)d_in[12];
  const float* gamma = (const float*)d_in[13];
  const float* beta  = (const float*)d_in[14];
  const float* w1    = (const float*)d_in[15];
  const float* b1    = (const float*)d_in[16];
  const float* w2    = (const float*)d_in[17];
  const float* b2    = (const float*)d_in[18];
  const int* src = eidx;
  const int* dst = eidx + Ee;

  char* wsb = (char*)d_ws; size_t off = 0;
  auto alloc = [&](size_t bytes) -> void* {
    void* p = wsb + off; off = (off + bytes + 255) & ~(size_t)255; return p;
  };
  float* x   = (float*)alloc((size_t)Nn * Hh * 4);
  float* h   = (float*)alloc((size_t)Nn * Hh * 4);
  float* S   = (float*)alloc((size_t)Nn * Rr * Hh * 4);
  float* inv = (float*)alloc((size_t)Nn * Rr * 4);
  unsigned short* Bpack = (unsigned short*)alloc((size_t)8 * KCHUNKS * 32 * 16 * 2);
  float* stats = (float*)alloc(2 * Hh * 4);
  float* psum  = (float*)alloc((size_t)Gg * Hh * 4);
  float* pmax  = (float*)alloc((size_t)Gg * Hh * 4);
  float* gcnt  = (float*)alloc((size_t)Gg * 4);
  float* ghid  = (float*)alloc((size_t)Gg * Hh * 4);

  const int NH_BLOCKS = (Nn * Hh) / 256;   // 50000 (exact)

  encode_kernel<<<NH_BLOCKS, 256, 0, stream>>>(x_ids, nfeat, emb, featw, featb, virt, x);
  fill_f32<<<1024, 256, 0, stream>>>(inv, 0.f, (long)Nn * Rr);
  count_edges<<<(Ee + 255) / 256, 256, 0, stream>>>(dst, etype, inv);
  invert_counts<<<(Nn * Rr + 255) / 256, 256, 0, stream>>>(inv);

  for (int l = 0; l < Ll; ++l) {
    build_bpack<<<(8 * KCHUNKS * 32 * 16) / 256, 256, 0, stream>>>(bases, comp, root, l, Bpack);
    fill_f32<<<4096, 256, 0, stream>>>(S, 0.f, (long)Nn * Rr * Hh);
    scatter_edges<<<Ee / 8, 256, 0, stream>>>(x, src, dst, etype, S);
    rgcn_gemm<<<(Nn + 15) / 16, 256, 0, stream>>>(S, inv, x, Bpack, rbias + l * Hh, h);
    fill_f32<<<1, 256, 0, stream>>>(stats, 0.f, 2 * Hh);
    bn_accum<<<(Nn + 127) / 128, 256, 0, stream>>>(h, stats);
    bn_update<<<NH_BLOCKS, 256, 0, stream>>>(h, stats, gamma + l * Hh, beta + l * Hh, x);
  }

  fill_f32<<<64, 256, 0, stream>>>(psum, 0.f, (long)Gg * Hh);
  fill_f32<<<64, 256, 0, stream>>>(pmax, -3.402823466e38f, (long)Gg * Hh);
  fill_f32<<<1, 128, 0, stream>>>(gcnt, 0.f, Gg);
  pool_kernel<<<NH_BLOCKS, 256, 0, stream>>>(x, batch, psum, pmax);
  graph_counts<<<(Nn + 255) / 256, 256, 0, stream>>>(batch, gcnt);
  mlp1_kernel<<<(Gg * Hh) / 256, 256, 0, stream>>>(psum, pmax, gcnt, w1, b1, ghid);
  mlp2_kernel<<<1, 256, 0, stream>>>(ghid, w2, b2, (float*)d_out);
}
